// Encoder_2207613190317
// MI455X (gfx1250) — matrix operand order repaired and checked by reading, not compiled, not run on hardware
//
#include <hip/hip_runtime.h>
#include <hip/hip_bf16.h>
#include <stdint.h>

typedef __bf16 bf16_t;
typedef __attribute__((ext_vector_type(16))) __bf16 v16bf;
typedef __attribute__((ext_vector_type(8)))  __bf16 bf16x8;
typedef __attribute__((ext_vector_type(8)))  float  v8f;

static __device__ __forceinline__ int imin(int a, int b) { return a < b ? a : b; }

// ---------------------------------------------------------------------------
// Generic batched GEMM:  C[m,n] = sum_k A[m,k] * B[n,k]   (B stored [N,K])
// A is bf16 (AF32=false) or fp32 converted in-register (AF32=true).
// Block = 256 thr = 8 waves; block tile 128x128; wave tile 32x64 (2x4 WMMA).
// Epilogue: +bias, +residual, relu, stores fp32 / bf16 / transposed-bf16.
// ---------------------------------------------------------------------------
template <bool AF32>
__global__ __launch_bounds__(256)
void gemm_bf16_wmma(const void* __restrict__ Aab, const bf16_t* __restrict__ B,
                    const float* __restrict__ bias, const float* __restrict__ resid,
                    float* __restrict__ Cf, bf16_t* __restrict__ Cb, bf16_t* __restrict__ CbT,
                    int M, int N, int K, int lda, int ldb, int ldc,
                    long long sAo, long long sAi, long long sBo, long long sBi,
                    long long sCo, long long sCi, int innerB, int relu)
{
    const int lane = threadIdx.x & 31;
    const int wave = threadIdx.x >> 5;
    const int half = lane >> 4;
    const int l15  = lane & 15;

    const int z  = blockIdx.z;
    const int zo = z / innerB;
    const int zi = z - zo * innerB;
    const long long offA = (long long)zo * sAo + (long long)zi * sAi;
    const long long offB = (long long)zo * sBo + (long long)zi * sBi;
    const long long offC = (long long)zo * sCo + (long long)zi * sCi;

    const int m0 = blockIdx.x * 128 + (wave >> 1) * 32;
    const int n0 = blockIdx.y * 128 + (wave & 1) * 64;

    v8f acc[2][4] = {};

    const float*  Af = (const float*)Aab;
    const bf16_t* Ab = (const bf16_t*)Aab;

    for (int kk = 0; kk < K; kk += 32) {
        // A fragments: 16x32 bf16, k(j) = kk + (j>>3)*16 + half*8 + (j&7)
        v16bf af[2];
        #pragma unroll
        for (int i = 0; i < 2; ++i) {
            const int m = imin(m0 + i * 16 + l15, M - 1);
            const long long base = offA + (long long)m * lda + kk + half * 8;
            union { v16bf v; bf16x8 h[2]; } u;
            if (AF32) {
                #pragma unroll
                for (int e = 0; e < 8; ++e) {
                    u.v[e]     = (bf16_t)Af[base + e];
                    u.v[e + 8] = (bf16_t)Af[base + 16 + e];
                }
            } else {
                u.h[0] = *(const bf16x8*)(Ab + base);
                u.h[1] = *(const bf16x8*)(Ab + base + 16);
            }
            af[i] = u.v;
        }
        // B fragments: 32x16 bf16, k(j) = kk + (j>>3)*16 + half*8 + (j&7)
        v16bf bfr[4];
        #pragma unroll
        for (int j = 0; j < 4; ++j) {
            const int n = imin(n0 + j * 16 + l15, N - 1);
            const bf16_t* bp = B + offB + (long long)n * ldb + kk + half * 8;
            union { v16bf v; bf16x8 h[2]; } u;
            u.h[0] = *(const bf16x8*)bp;
            u.h[1] = *(const bf16x8*)(bp + 16);
            bfr[j] = u.v;
        }
        #pragma unroll
        for (int i = 0; i < 2; ++i)
            #pragma unroll
            for (int j = 0; j < 4; ++j)
                acc[i][j] = __builtin_amdgcn_wmma_f32_16x16x32_bf16(
                    false, af[i], false, bfr[j], (short)0, acc[i][j], false, false);
    }

    // Epilogue. C/D layout: VGPR r -> m = tile + half*8 + r, n = tile + l15
    #pragma unroll
    for (int i = 0; i < 2; ++i) {
        const int mbase = m0 + i * 16 + half * 8;
        #pragma unroll
        for (int j = 0; j < 4; ++j) {
            const int n = n0 + j * 16 + l15;
            if (n >= N) continue;
            const float bv = bias ? bias[n] : 0.0f;
            #pragma unroll
            for (int r = 0; r < 8; ++r) {
                const int m = mbase + r;
                if (m >= M) continue;
                float c = acc[i][j][r] + bv;
                const long long ci = offC + (long long)m * ldc + n;
                if (resid) c += resid[ci];
                if (relu)  c = fmaxf(c, 0.0f);
                if (Cf) Cf[ci] = c;
                if (Cb) Cb[ci] = (bf16_t)c;
                if (CbT) {  // [B=16,S=512]x[H=8,D=64] -> [b,h,d,s]
                    const long long ti =
                        ((((long long)(m >> 9)) * 8 + (n >> 6)) * 64 + (n & 63)) * 512 + (m & 511);
                    CbT[ti] = (bf16_t)c;
                }
            }
        }
    }
}

// ---------------------------------------------------------------------------
// In-place masked softmax over rows of 512, one wave per row.
// Row layout: [b][h][q][k]; mask where tokens[b][k]==0 -> -1e9 (after *scale).
// ---------------------------------------------------------------------------
__global__ __launch_bounds__(256)
void softmax_mask(float* __restrict__ probs, const int* __restrict__ tokens, float scale)
{
    const int gw   = (int)((blockIdx.x * 256 + threadIdx.x) >> 5);
    const int lane = threadIdx.x & 31;
    const int b    = gw >> 12;            // gw = ((b*8+h)*512+q); b = gw/(8*512)
    float* row = probs + (long long)gw * 512;
    const int* tok = tokens + b * 512;

    float v[16];
    float mx = -3.0e38f;
    #pragma unroll
    for (int t = 0; t < 16; ++t) {
        const int k = lane + t * 32;
        float s = row[k];
        s = (tok[k] == 0) ? -1.0e9f : s * scale;
        v[t] = s;
        mx = fmaxf(mx, s);
    }
    #pragma unroll
    for (int m = 16; m >= 1; m >>= 1) mx = fmaxf(mx, __shfl_xor(mx, m, 32));
    float sum = 0.0f;
    #pragma unroll
    for (int t = 0; t < 16; ++t) { v[t] = __expf(v[t] - mx); sum += v[t]; }
    #pragma unroll
    for (int m = 16; m >= 1; m >>= 1) sum += __shfl_xor(sum, m, 32);
    const float inv = 1.0f / sum;
    #pragma unroll
    for (int t = 0; t < 16; ++t) row[lane + t * 32] = v[t] * inv;
}

// ---------------------------------------------------------------------------
// LayerNorm over rows of 512, one wave per row; writes fp32 + bf16 copies.
// ---------------------------------------------------------------------------
__global__ __launch_bounds__(256)
void layernorm_rows(const float* __restrict__ in, const float* __restrict__ g,
                    const float* __restrict__ be,
                    float* __restrict__ outf, bf16_t* __restrict__ outb)
{
    const int gw   = (int)((blockIdx.x * 256 + threadIdx.x) >> 5);
    const int lane = threadIdx.x & 31;
    const float* row = in + (long long)gw * 512;

    float v[16];
    float s = 0.0f;
    #pragma unroll
    for (int t = 0; t < 16; ++t) { v[t] = row[lane + t * 32]; s += v[t]; }
    #pragma unroll
    for (int m = 16; m >= 1; m >>= 1) s += __shfl_xor(s, m, 32);
    const float mu = s * (1.0f / 512.0f);

    float q = 0.0f;
    #pragma unroll
    for (int t = 0; t < 16; ++t) { const float d = v[t] - mu; q += d * d; }
    #pragma unroll
    for (int m = 16; m >= 1; m >>= 1) q += __shfl_xor(q, m, 32);
    const float rstd = rsqrtf(q * (1.0f / 512.0f) + 1e-5f);

    #pragma unroll
    for (int t = 0; t < 16; ++t) {
        const int n = lane + t * 32;
        const float y = (v[t] - mu) * rstd * g[n] + be[n];
        const long long idx = (long long)gw * 512 + n;
        outf[idx] = y;
        outb[idx] = (bf16_t)y;
    }
}

// ---------------------------------------------------------------------------
// x = emb[token] + sinusoid pos-enc(pos = s+1); writes fp32 + bf16 copies.
// ---------------------------------------------------------------------------
__global__ __launch_bounds__(256)
void embed_pos(const int* __restrict__ tokens, const float* __restrict__ emb,
               float* __restrict__ xf, bf16_t* __restrict__ xb)
{
    const int gw   = (int)((blockIdx.x * 256 + threadIdx.x) >> 5);   // b*512+s
    const int lane = threadIdx.x & 31;
    const int s    = gw & 511;
    const int tok  = tokens[gw];
    const float* e = emb + (long long)tok * 512;
    const float p  = (float)(s + 1);

    #pragma unroll
    for (int t = 0; t < 16; ++t) {
        const int j = lane + t * 32;
        const float expo  = (float)(j & ~1) / 512.0f;           // 2*floor(j/2)/d
        const float angle = p * powf(10000.0f, -expo);
        const float pe    = (j & 1) ? cosf(angle) : sinf(angle);
        const float val   = e[j] + pe;
        const long long idx = (long long)gw * 512 + j;
        xf[idx] = val;
        xb[idx] = (bf16_t)val;
    }
}

__global__ __launch_bounds__(256)
void cvt_f32_bf16(const float* __restrict__ in, bf16_t* __restrict__ out, long long n)
{
    long long i = (long long)blockIdx.x * 256 + threadIdx.x;
    const long long stride = (long long)gridDim.x * 256;
    for (; i < n; i += stride) out[i] = (bf16_t)in[i];
}

// ---------------------------------------------------------------------------
extern "C" void kernel_launch(void* const* d_in, const int* in_sizes, int n_in,
                              void* d_out, int out_size, void* d_ws, size_t ws_size,
                              hipStream_t stream)
{
    (void)in_sizes; (void)n_in; (void)out_size; (void)ws_size;

    const int*   tok = (const int*)  d_in[0];
    const float* emb = (const float*)d_in[1];
    const float* Wq  = (const float*)d_in[2];
    const float* bq  = (const float*)d_in[3];
    const float* Wk  = (const float*)d_in[4];
    const float* bk  = (const float*)d_in[5];
    const float* Wv  = (const float*)d_in[6];
    const float* bv  = (const float*)d_in[7];
    const float* Wo  = (const float*)d_in[8];
    const float* bo  = (const float*)d_in[9];
    const float* g1  = (const float*)d_in[10];
    const float* b1  = (const float*)d_in[11];
    const float* Wc1 = (const float*)d_in[12];
    const float* bc1 = (const float*)d_in[13];
    const float* Wc2 = (const float*)d_in[14];
    const float* bc2 = (const float*)d_in[15];
    const float* g2  = (const float*)d_in[16];
    const float* b2  = (const float*)d_in[17];

    float* outX = (float*)d_out;                       // [16,512,512] fp32
    float* outW = outX + (long long)16 * 512 * 512;    // [6,16,8,512,512] fp32

    // ---- workspace carve-up (all chunks 256B aligned) ----
    char* p = (char*)d_ws;
    auto take = [&](size_t bytes) -> char* {
        char* r = p;
        p += (bytes + 255) & ~(size_t)255;
        return r;
    };
    const long long W2 = 512ll * 512;            // per-layer square weight
    bf16_t* wqb  = (bf16_t*)take(6ll * W2 * 2);
    bf16_t* wkb  = (bf16_t*)take(6ll * W2 * 2);
    bf16_t* wvb  = (bf16_t*)take(6ll * W2 * 2);
    bf16_t* wob  = (bf16_t*)take(6ll * W2 * 2);
    bf16_t* wc1b = (bf16_t*)take(6ll * 2048 * 512 * 2);
    bf16_t* wc2b = (bf16_t*)take(6ll * 512 * 2048 * 2);
    bf16_t* qb   = (bf16_t*)take(8192ll * 512 * 2);
    bf16_t* kb   = (bf16_t*)take(8192ll * 512 * 2);
    bf16_t* vb   = (bf16_t*)take(8192ll * 512 * 2);
    bf16_t* vtb  = (bf16_t*)take(8192ll * 512 * 2);   // V transposed [b,h,d,s]
    bf16_t* hb   = qb;  // FFN hidden [8192,2048] bf16 aliases dead q/k/v/vt (32MB)
    bf16_t* ctxb = (bf16_t*)take(8192ll * 512 * 2);
    float*  xf   = (float*)take(8192ll * 512 * 4);
    bf16_t* xb   = (bf16_t*)take(8192ll * 512 * 2);
    float*  tmpf = (float*)take(8192ll * 512 * 4);

    // ---- weight conversion to bf16 (one-time work per call, small) ----
    cvt_f32_bf16<<<2048, 256, 0, stream>>>(Wq,  wqb,  6ll * W2);
    cvt_f32_bf16<<<2048, 256, 0, stream>>>(Wk,  wkb,  6ll * W2);
    cvt_f32_bf16<<<2048, 256, 0, stream>>>(Wv,  wvb,  6ll * W2);
    cvt_f32_bf16<<<2048, 256, 0, stream>>>(Wo,  wob,  6ll * W2);
    cvt_f32_bf16<<<4096, 256, 0, stream>>>(Wc1, wc1b, 6ll * 2048 * 512);
    cvt_f32_bf16<<<4096, 256, 0, stream>>>(Wc2, wc2b, 6ll * 512 * 2048);

    // ---- embedding + positional encoding ----
    embed_pos<<<1024, 256, 0, stream>>>(tok, emb, xf, xb);

    const float scale = 0.125f;  // 1/sqrt(64)

    for (int l = 0; l < 6; ++l) {
        // Q, K, V projections: [8192,512] = x @ W^T + b   (bf16 out)
        gemm_bf16_wmma<false><<<dim3(64, 4, 1), 256, 0, stream>>>(
            xb, wqb + l * W2, bq + l * 512, nullptr, nullptr, qb, nullptr,
            8192, 512, 512, 512, 512, 512, 0, 0, 0, 0, 0, 0, 1, 0);
        gemm_bf16_wmma<false><<<dim3(64, 4, 1), 256, 0, stream>>>(
            xb, wkb + l * W2, bk + l * 512, nullptr, nullptr, kb, nullptr,
            8192, 512, 512, 512, 512, 512, 0, 0, 0, 0, 0, 0, 1, 0);
        gemm_bf16_wmma<false><<<dim3(64, 4, 1), 256, 0, stream>>>(
            xb, wvb + l * W2, bv + l * 512, nullptr, nullptr, vb, vtb,
            8192, 512, 512, 512, 512, 512, 0, 0, 0, 0, 0, 0, 1, 0);

        // scores[b,h,q,k] = Q.K^T  -> fp32 straight into d_out attn slot
        float* attn_l = outW + (long long)l * 16 * 8 * 512 * 512;
        gemm_bf16_wmma<false><<<dim3(4, 4, 128), 256, 0, stream>>>(
            qb, kb, nullptr, nullptr, attn_l, nullptr, nullptr,
            512, 512, 64, 512, 512, 512,
            262144, 64, 262144, 64, 2097152, 262144, 8, 0);

        // in-place masked softmax (scale + pad mask + softmax)
        softmax_mask<<<8192, 256, 0, stream>>>(attn_l, tok, scale);

        // ctx[b,s,h*64+d] = probs @ V   (A fp32 from d_out, B = V^T bf16)
        gemm_bf16_wmma<true><<<dim3(4, 1, 128), 256, 0, stream>>>(
            attn_l, vtb, nullptr, nullptr, nullptr, ctxb, nullptr,
            512, 64, 512, 512, 512, 512,
            2097152, 262144, 262144, 32768, 262144, 64, 8, 0);

        // out-proj + bias + residual -> tmp; then LayerNorm -> x
        gemm_bf16_wmma<false><<<dim3(64, 4, 1), 256, 0, stream>>>(
            ctxb, wob + l * W2, bo + l * 512, xf, tmpf, nullptr, nullptr,
            8192, 512, 512, 512, 512, 512, 0, 0, 0, 0, 0, 0, 1, 0);
        layernorm_rows<<<1024, 256, 0, stream>>>(tmpf, g1 + l * 512, b1 + l * 512, xf, xb);

        // FFN: h = relu(x @ Wc1^T + bc1) (bf16)
        gemm_bf16_wmma<false><<<dim3(64, 16, 1), 256, 0, stream>>>(
            xb, wc1b + l * 2048ll * 512, bc1 + l * 2048, nullptr, nullptr, hb, nullptr,
            8192, 2048, 512, 512, 512, 2048, 0, 0, 0, 0, 0, 0, 1, 1);
        // x = LN(h @ Wc2^T + bc2 + x)
        gemm_bf16_wmma<false><<<dim3(64, 4, 1), 256, 0, stream>>>(
            hb, wc2b + l * 512ll * 2048, bc2 + l * 512, xf, tmpf, nullptr, nullptr,
            8192, 512, 2048, 2048, 2048, 512, 0, 0, 0, 0, 0, 0, 1, 0);
        layernorm_rows<<<1024, 256, 0, stream>>>(tmpf, g2 + l * 512, b2 + l * 512, xf, xb);
    }

    // final x -> d_out
    hipMemcpyAsync(outX, xf, (size_t)16 * 512 * 512 * 4,
                   hipMemcpyDeviceToDevice, stream);
}
